// GlobalPointer_25494925869453
// MI455X (gfx1250) — matrix operand order repaired and checked by reading, not compiled, not run on hardware
//
#include <hip/hip_runtime.h>
#include <hip/hip_bf16.h>

#define HEADS 12
#define HEAD_SIZE 64
#define HIDDEN 1024
#define BATCH 32
#define SEQ 512
#define NQKV (HEADS * 2 * HEAD_SIZE)   // 1536
#define MTOT (BATCH * SEQ)             // 16384
#define INF_ 1000000000000.0f

typedef __attribute__((ext_vector_type(16))) _Float16 v16h;
typedef __attribute__((ext_vector_type(8)))  float    v8f;

union F16Frag { unsigned u[8]; v16h v; };

// ---------------------------------------------------------------------------
// CDNA5 async memory->LDS copy, GVS form (CDNA5 ISA):
//   global addr = SGPR64 base + VGPR32 byte-offset + INST_OFFSET
//   LDS addr    = VGPR32 lds-offset + INST_OFFSET
// The uniform K-tile advance rides the SGPR base (SALU); per-thread offsets
// stay in one fixed VGPR each. LDS byte address = low 32 bits of generic ptr.
// ---------------------------------------------------------------------------
#define ASYNC_B128_S(lds, voff, sbase, SUFFIX)                            \
  asm volatile("global_load_async_to_lds_b128 %0, %1, %2" SUFFIX          \
               :: "v"((unsigned)(size_t)(const void*)(lds)),              \
                  "v"((unsigned)(voff)),                                  \
                  "s"((unsigned long long)(size_t)(const void*)(sbase))   \
               : "memory")

__device__ __forceinline__ void wait_async0() {
  asm volatile("s_wait_asynccnt 0" ::: "memory");
}

// ---------------------------------------------------------------------------
// Fragment loaders from LDS (wave32 WMMA f16 layouts, CDNA5 ISA)
// ---------------------------------------------------------------------------
__device__ __forceinline__ v16h load_frag_A(const _Float16* base, int stride,
                                            int row, int lane) {
  F16Frag f;
  int r  = row + (lane & 15);
  int kh = (lane >> 4) * 8;
  const unsigned* p = (const unsigned*)(base + r * stride);
#pragma unroll
  for (int j = 0; j < 4; ++j) f.u[j] = p[j + kh];
#pragma unroll
  for (int j = 0; j < 4; ++j) f.u[4 + j] = p[4 + j + kh];
  return f.v;
}

__device__ __forceinline__ v16h load_frag_B(const _Float16* base, int stride,
                                            int col, int lane) {
  F16Frag f;
  int n  = col + (lane & 15);
  int kh = (lane >> 4) * 8;
  const unsigned* p = (const unsigned*)(base + n * stride);
#pragma unroll
  for (int v = 0; v < 8; ++v) f.u[v] = p[v + kh];
  return f.v;
}

// ---------------------------------------------------------------------------
// fp32 -> f16 converts
// ---------------------------------------------------------------------------
__global__ void k_cvt_inputs(const float* __restrict__ in,
                             _Float16* __restrict__ out, int n) {
  int i = (blockIdx.x * blockDim.x + threadIdx.x) * 8;
  if (i >= n) return;
  const float4* p = (const float4*)(in + i);
  float4 a = p[0], b = p[1];
  _Float16 h[8] = {(_Float16)a.x, (_Float16)a.y, (_Float16)a.z, (_Float16)a.w,
                   (_Float16)b.x, (_Float16)b.y, (_Float16)b.z, (_Float16)b.w};
  *(uint4*)(out + i) = *(const uint4*)h;
}

__global__ void k_cvt_w(const float* __restrict__ W, _Float16* __restrict__ Wt) {
  int idx = blockIdx.x * blockDim.x + threadIdx.x;
  if (idx >= HIDDEN * NQKV) return;
  int k = idx / NQKV, n = idx % NQKV;
  Wt[n * HIDDEN + k] = (_Float16)W[idx];
}

// ---------------------------------------------------------------------------
// GEMM1: qkv[m][n] = A16[m][k]*Wt16[n][k] + bias[n]  (M=16384,N=1536,K=1024)
// 128x128 tile, 8 waves (4x2), 32x64/wave. b128 async-to-LDS double buffer.
// ---------------------------------------------------------------------------
#define BT1  128
#define KT1  32
#define LDA1 40                 // halfs; 80B rows -> b128-aligned staging
#define SZ1  (BT1 * LDA1)       // per-matrix tile, halfs

__global__ __launch_bounds__(256) void k_gemm_qkv(
    const _Float16* __restrict__ A, const _Float16* __restrict__ Wt,
    const float* __restrict__ bias, _Float16* __restrict__ qkv) {
  __shared__ _Float16 sm[2 * 2 * SZ1];    // [buf][A/B][128*40]

  const int bidm = blockIdx.x / (NQKV / BT1);
  const int bidn = blockIdx.x % (NQKV / BT1);
  const int m0 = bidm * BT1, n0 = bidn * BT1;
  const int t = threadIdx.x, lane = t & 31, wave = t >> 5;
  const int wr = wave & 3, wc = wave >> 2;

  v8f acc[2][4];
#pragma unroll
  for (int a = 0; a < 2; ++a)
#pragma unroll
    for (int b = 0; b < 4; ++b) acc[a][b] = (v8f)0.0f;

  const int row = t >> 1, seg = t & 1;    // 128 rows x 2 segments of 16 halfs
  const unsigned voffA = (unsigned)(((m0 + row) * HIDDEN + seg * 16) * 2);
  const unsigned voffB = (unsigned)(((n0 + row) * HIDDEN + seg * 16) * 2);
  _Float16* stA[2] = {&sm[0 * 2 * SZ1 + row * LDA1 + seg * 16],
                      &sm[1 * 2 * SZ1 + row * LDA1 + seg * 16]};
  _Float16* stB[2] = {&sm[0 * 2 * SZ1 + SZ1 + row * LDA1 + seg * 16],
                      &sm[1 * 2 * SZ1 + SZ1 + row * LDA1 + seg * 16]};

  auto issue_tile = [&](int bufi, int kt) {
    const _Float16* sa = A + kt;          // uniform -> SGPR pair
    const _Float16* sb = Wt + kt;
    ASYNC_B128_S(stA[bufi], voffA, sa, "");
    ASYNC_B128_S(stA[bufi], voffA, sa, " offset:16");
    ASYNC_B128_S(stB[bufi], voffB, sb, "");
    ASYNC_B128_S(stB[bufi], voffB, sb, " offset:16");
  };

  // warm WGP$ for tile 1 while tile 0's async copies are in flight
  __builtin_prefetch(A + voffA / 2 + KT1, 0, 3);
  __builtin_prefetch(Wt + voffB / 2 + KT1, 0, 3);
  issue_tile(0, 0);
  wait_async0();
  __syncthreads();

  int buf = 0;
#pragma unroll 2
  for (int kt = 0; kt < HIDDEN; kt += KT1) {
    if (kt + KT1 < HIDDEN)
      issue_tile(buf ^ 1, kt + KT1);      // hidden behind the WMMAs

    const _Float16* bA = &sm[buf * 2 * SZ1];
    const _Float16* bB = &sm[buf * 2 * SZ1 + SZ1];
    F16Frag af[2], bf[4];
#pragma unroll
    for (int a = 0; a < 2; ++a)
      af[a].v = load_frag_A(bA, LDA1, wr * 32 + a * 16, lane);
#pragma unroll
    for (int b = 0; b < 4; ++b)
      bf[b].v = load_frag_B(bB, LDA1, wc * 64 + b * 16, lane);
#pragma unroll
    for (int a = 0; a < 2; ++a)
#pragma unroll
      for (int b = 0; b < 4; ++b)
        acc[a][b] = __builtin_amdgcn_wmma_f32_16x16x32_f16(
            false, af[a].v, false, bf[b].v, (short)0, acc[a][b], false, false);

    wait_async0();
    __syncthreads();
    buf ^= 1;
  }

  // epilogue: +bias, cvt f16. C layout: N = lane&15, M = r + 8*(lane>>4)
  const int nlane = lane & 15, moff = (lane >> 4) * 8;
#pragma unroll
  for (int a = 0; a < 2; ++a)
#pragma unroll
    for (int b = 0; b < 4; ++b) {
      int n = n0 + wc * 64 + b * 16 + nlane;
      float bv = bias[n];
#pragma unroll
      for (int r = 0; r < 8; ++r) {
        int m = m0 + wr * 32 + a * 16 + r + moff;
        qkv[(size_t)m * NQKV + n] = (_Float16)(acc[a][b][r] + bv);
      }
    }
}

// ---------------------------------------------------------------------------
// RoPE: split qkv -> Q/K in [b][h][s][d] f16 (d contiguous = GEMM2 K-dim)
// ---------------------------------------------------------------------------
__global__ void k_rope(const _Float16* __restrict__ qkv,
                       _Float16* __restrict__ Q, _Float16* __restrict__ K) {
  int idx = blockIdx.x * blockDim.x + threadIdx.x;
  if (idx >= BATCH * SEQ * HEADS * (HEAD_SIZE / 2)) return;
  int i = idx & 31;
  int h = (idx >> 5) % HEADS;
  int m = (idx / (32 * HEADS)) % SEQ;
  int b = idx / (32 * HEADS * SEQ);

  size_t rowq = (size_t)(b * SEQ + m) * NQKV + h * (2 * HEAD_SIZE);
  float q1 = (float)qkv[rowq + 2 * i];
  float q2 = (float)qkv[rowq + 2 * i + 1];
  float k1 = (float)qkv[rowq + HEAD_SIZE + 2 * i];
  float k2 = (float)qkv[rowq + HEAD_SIZE + 2 * i + 1];

  float inv_freq = expf(-(float)(2 * i) * 0.14391156831212787f); // 10000^(-2i/64)
  float ang = (float)m * inv_freq;
  float s = sinf(ang), c = cosf(ang);

  size_t orow = (size_t)((b * HEADS + h) * SEQ + m) * HEAD_SIZE;
  Q[orow + 2 * i]     = (_Float16)(q1 * c - q2 * s);
  Q[orow + 2 * i + 1] = (_Float16)(q1 * s + q2 * c);
  K[orow + 2 * i]     = (_Float16)(k1 * c - k2 * s);
  K[orow + 2 * i + 1] = (_Float16)(k1 * s + k2 * c);
}

// ---------------------------------------------------------------------------
// GEMM2: logits[b,h,m,n] = sum_d Q[b,h,m,d]*K[b,h,n,d]; mask + scale fused.
// Per (b,h): M=N=512, K=64 (2 WMMA k-steps, fully staged via b128 asyncs).
// ---------------------------------------------------------------------------
#define LDA2 72                 // halfs; 144B rows -> b128-aligned staging
#define SZ2  (128 * LDA2)

__global__ __launch_bounds__(256) void k_attn(
    const _Float16* __restrict__ Q, const _Float16* __restrict__ K,
    const int* __restrict__ am, float* __restrict__ out) {
  __shared__ _Float16 sm[2 * SZ2];        // [Q tile][K tile]

  const int bh = blockIdx.y;              // 0..383
  const int b = bh / HEADS;
  const int tm0 = (blockIdx.x & 3) * 128, tn0 = (blockIdx.x >> 2) * 128;
  const int t = threadIdx.x, lane = t & 31, wave = t >> 5;
  const int wr = wave & 3, wc = wave >> 2;

  // async-stage Q and K tiles (128 rows x 64 halfs each; 64B per thread/tile)
  const int row = t >> 1, seg = t & 1;
  const unsigned voff = (unsigned)((row * HEAD_SIZE + seg * 32) * 2);
  const _Float16* sqb = Q + ((size_t)bh * SEQ + tm0) * HEAD_SIZE;  // uniform
  const _Float16* skb = K + ((size_t)bh * SEQ + tn0) * HEAD_SIZE;  // uniform
  _Float16* sq = &sm[row * LDA2 + seg * 32];
  _Float16* sk = &sm[SZ2 + row * LDA2 + seg * 32];
  ASYNC_B128_S(sq, voff, sqb, "");
  ASYNC_B128_S(sq, voff, sqb, " offset:16");
  ASYNC_B128_S(sq, voff, sqb, " offset:32");
  ASYNC_B128_S(sq, voff, sqb, " offset:48");
  ASYNC_B128_S(sk, voff, skb, "");
  ASYNC_B128_S(sk, voff, skb, " offset:16");
  ASYNC_B128_S(sk, voff, skb, " offset:32");
  ASYNC_B128_S(sk, voff, skb, " offset:48");
  wait_async0();
  __syncthreads();

  v8f acc[2][4];
#pragma unroll
  for (int a = 0; a < 2; ++a)
#pragma unroll
    for (int c = 0; c < 4; ++c) acc[a][c] = (v8f)0.0f;

#pragma unroll
  for (int ks = 0; ks < 2; ++ks) {
    F16Frag af[2], bf[4];
#pragma unroll
    for (int a = 0; a < 2; ++a)
      af[a].v = load_frag_A(sm + ks * 32, LDA2, wr * 32 + a * 16, lane);
#pragma unroll
    for (int c = 0; c < 4; ++c)
      bf[c].v = load_frag_B(sm + SZ2 + ks * 32, LDA2, wc * 64 + c * 16, lane);
#pragma unroll
    for (int a = 0; a < 2; ++a)
#pragma unroll
      for (int c = 0; c < 4; ++c)
        acc[a][c] = __builtin_amdgcn_wmma_f32_16x16x32_f16(
            false, af[a].v, false, bf[c].v, (short)0, acc[a][c], false, false);
  }

  // epilogue: pad mask + tril(k=-1) mask (n<m) + scale 1/sqrt(64)
  const int nlane = lane & 15, moff = (lane >> 4) * 8;
  float amn[4];
#pragma unroll
  for (int c = 0; c < 4; ++c)
    amn[c] = (float)am[b * SEQ + tn0 + wc * 64 + c * 16 + nlane];

#pragma unroll
  for (int a = 0; a < 2; ++a)
#pragma unroll
    for (int r = 0; r < 8; ++r) {
      int m = tm0 + wr * 32 + a * 16 + r + moff;
      float amm = (float)am[b * SEQ + m];
      size_t obase = ((size_t)bh * SEQ + m) * SEQ;
#pragma unroll
      for (int c = 0; c < 4; ++c) {
        int n = tn0 + wc * 64 + c * 16 + nlane;
        float v = acc[a][c][r];
        v -= (1.0f - amn[c] * amm) * INF_;  // pad mask (no-op for all-ones)
        if (n < m) v -= INF_;               // strict lower-triangle mask
        out[obase + n] = v * 0.125f;
      }
    }
}

// ---------------------------------------------------------------------------
extern "C" void kernel_launch(void* const* d_in, const int* in_sizes, int n_in,
                              void* d_out, int out_size, void* d_ws, size_t ws_size,
                              hipStream_t stream) {
  const float* inputs = (const float*)d_in[0];
  const float* W      = (const float*)d_in[1];
  const float* bias   = (const float*)d_in[2];
  const int*   am     = (const int*)d_in[3];
  float* out = (float*)d_out;

  char* ws = (char*)d_ws;
  _Float16* A16   = (_Float16*)(ws);                     // 32 MiB
  _Float16* Wt16  = (_Float16*)(ws + 33554432);          //  3 MiB
  _Float16* QKV16 = (_Float16*)(ws + 36700160);          // 48 MiB
  _Float16* Q16   = (_Float16*)(ws + 87031808);          // 24 MiB
  _Float16* K16   = (_Float16*)(ws + 112197632);         // 24 MiB

  int nA = MTOT * HIDDEN;
  k_cvt_inputs<<<(nA / 8 + 255) / 256, 256, 0, stream>>>(inputs, A16, nA);
  k_cvt_w<<<(HIDDEN * NQKV + 255) / 256, 256, 0, stream>>>(W, Wt16);
  k_gemm_qkv<<<(MTOT / BT1) * (NQKV / BT1), 256, 0, stream>>>(A16, Wt16, bias, QKV16);
  k_rope<<<(BATCH * SEQ * HEADS * (HEAD_SIZE / 2) + 255) / 256, 256, 0, stream>>>(QKV16, Q16, K16);
  dim3 g2(16, BATCH * HEADS);
  k_attn<<<g2, 256, 0, stream>>>(Q16, K16, am, out);
}
